// GPTSelfAttentionV1_10514079941360
// MI455X (gfx1250) — matrix operand order repaired and checked by reading, not compiled, not run on hardware
//
#include <hip/hip_runtime.h>
#include <hip/hip_bf16.h>
#include <math.h>

// ---------------------------------------------------------------------------
// GPT self-attention, B=4, S=2048, E=2048, D=2048, fp32 in/out.
// Compute-bound (~280 GF matmul vs ~190MB HBM) -> bf16 WMMA with f32 accum.
// Pipeline: cvt -> 3x proj GEMM -> V transpose -> QK^T GEMM (causal block
// skip) -> causal softmax (LDS row cache) -> P*V GEMM (K capped) -> f32 out.
// GEMM: 128x128 block, 4 waves, 64x64 wave tile = 16 WMMA / 16 ds_b128 per
// K-step, double-buffered LDS (40KB). Global->LDS staging uses
// GLOBAL_LOAD_ASYNC_TO_LDS_B128 (ASYNCcnt, no VGPR round-trip) when the
// toolchain exposes it; otherwise explicit SW pipelining places the
// load-wait AFTER the WMMA block so HBM latency hides behind matrix math.
// Workspace required: ~235 MB (guarded).
// ---------------------------------------------------------------------------

typedef __bf16 bf16_t;
typedef __attribute__((ext_vector_type(8)))  __bf16 v8bf;
typedef __attribute__((ext_vector_type(16))) __bf16 v16bf;
typedef __attribute__((ext_vector_type(8)))  float  v8f;

#if defined(__gfx1250__) && defined(__has_builtin)
#if __has_builtin(__builtin_amdgcn_global_load_async_to_lds_b128)
#define HAVE_ASYNC_LDS 1
#endif
#endif
#ifndef HAVE_ASYNC_LDS
#define HAVE_ASYNC_LDS 0
#endif

#if HAVE_ASYNC_LDS
// Builtin parameter 0 is a generic 'int4*' (per clang diagnostic); param 1 is
// the LDS destination -> AS(3)-qualified int4* (AS3->generic also implicit).
typedef int v4i_vs __attribute__((vector_size(16)));
typedef __attribute__((address_space(3))) v4i_vs as3_v4i;
__device__ inline void waitAsync0() {
#if __has_builtin(__builtin_amdgcn_s_wait_asynccnt)
  __builtin_amdgcn_s_wait_asynccnt(0);
#else
  asm volatile("s_wait_asynccnt 0x0" ::: "memory");
#endif
}
#endif

#define BM 128   // block tile M
#define BN 128   // block tile N
#define BK 32    // K step (matches 16x16x32 bf16 WMMA depth)
#define BKP 40   // padded LDS row (bf16); 80B keeps 16B alignment, kills bank conflicts
#define WM 64    // wave tile M (4 frags)
#define WN 64    // wave tile N (4 frags)

__device__ inline v16bf cat8(v8bf lo, v8bf hi) {
  return __builtin_shufflevector(lo, hi, 0,1,2,3,4,5,6,7,8,9,10,11,12,13,14,15);
}

// fp32 -> bf16 bulk convert, 8 elems/thread, 16B loads/stores
__global__ __launch_bounds__(256) void cvt_f32_bf16_kernel(
    const float* __restrict__ src, bf16_t* __restrict__ dst, int n8) {
  int i = blockIdx.x * 256 + threadIdx.x;
  if (i >= n8) return;
  const float4* s4 = (const float4*)src;
  float4 a = s4[2*i], b = s4[2*i + 1];
  v8bf o;
  o[0]=(bf16_t)a.x; o[1]=(bf16_t)a.y; o[2]=(bf16_t)a.z; o[3]=(bf16_t)a.w;
  o[4]=(bf16_t)b.x; o[5]=(bf16_t)b.y; o[6]=(bf16_t)b.z; o[7]=(bf16_t)b.w;
  *(v8bf*)&dst[(size_t)i * 8] = o;
}

// ---------------------------------------------------------------------------
// C[M,N] = A[M,K] * B[N,K]^T  (both operands K-major -> "NT" GEMM)
// OUT_BF16: 1 -> write bf16 to Cb, 0 -> write f32 to Cf
// CAUSAL_MODE: 0 none; 1 skip blocks fully above diagonal (scores GEMM);
//              2 cap K loop at m0+BM (output GEMM; P is zero past diagonal)
// 128 threads = 4 waves (2x2), wave tile 64x64 = 4x4 WMMA accumulators.
// ---------------------------------------------------------------------------
template<int OUT_BF16, int CAUSAL_MODE>
__global__ __launch_bounds__(128) void gemm_nt_kernel(
    const bf16_t* __restrict__ A, const bf16_t* __restrict__ B,
    float* __restrict__ Cf, bf16_t* __restrict__ Cb,
    int M, int N, int K,
    size_t aBatch, size_t bBatch, size_t cBatch)
{
  const int bz = blockIdx.z;
  A += (size_t)bz * aBatch;
  B += (size_t)bz * bBatch;

  const int m0 = blockIdx.y * BM;
  const int n0 = blockIdx.x * BN;
  if (CAUSAL_MODE == 1 && n0 > m0 + (BM - 1)) return;   // fully masked tile
  int kEnd = K;
  if (CAUSAL_MODE == 2) { int cap = m0 + BM; kEnd = cap < K ? cap : K; }

  __shared__ __align__(16) bf16_t sA[2][BM][BKP];
  __shared__ __align__(16) bf16_t sB[2][BN][BKP];

  const int tid   = threadIdx.x;
  const int wave  = tid >> 5;
  const int lane  = tid & 31;
  const int waveM = wave >> 1;            // 0..1
  const int waveN = wave & 1;             // 0..1
  const int wmB   = waveM * WM;
  const int wnB   = waveN * WN;
  const int lmod  = lane & 15;
  const int lhalf = lane >> 4;

  // global->LDS: each thread moves 4 rows x 16B per matrix per tile
  const int lr = tid >> 2;                // 0..31
  const int lc = (tid & 3) * 8;           // bf16 col offset 0/8/16/24

  v8f acc[4][4];
  #pragma unroll
  for (int i = 0; i < 4; ++i)
    #pragma unroll
    for (int j = 0; j < 4; ++j) acc[i][j] = {};

#if HAVE_ASYNC_LDS
  // CDNA5 async copy: memory -> LDS with no VGPR staging (ASYNCcnt tracked)
  auto loadTileAsync = [&](int kk, int buf) {
    #pragma unroll
    for (int r = 0; r < 4; ++r) {
      int row = lr + r * 32;
      __builtin_amdgcn_global_load_async_to_lds_b128(
          (v4i_vs*)&A[(size_t)(m0 + row) * K + kk + lc],
          (as3_v4i*)&sA[buf][row][lc], 0, 0);
      __builtin_amdgcn_global_load_async_to_lds_b128(
          (v4i_vs*)&B[(size_t)(n0 + row) * K + kk + lc],
          (as3_v4i*)&sB[buf][row][lc], 0, 0);
    }
  };
#else
  // SW-pipelined VGPR staging: loads issued before WMMA block, stores after
  auto loadG = [&](int kk, v8bf* ga, v8bf* gb) {
    #pragma unroll
    for (int r = 0; r < 4; ++r) {
      int row = lr + r * 32;
      ga[r] = *(const v8bf*)&A[(size_t)(m0 + row) * K + kk + lc];
      gb[r] = *(const v8bf*)&B[(size_t)(n0 + row) * K + kk + lc];
    }
  };
  auto storeL = [&](int buf, const v8bf* ga, const v8bf* gb) {
    #pragma unroll
    for (int r = 0; r < 4; ++r) {
      int row = lr + r * 32;
      *(v8bf*)&sA[buf][row][lc] = ga[r];
      *(v8bf*)&sB[buf][row][lc] = gb[r];
    }
  };
#endif

  // ISA 16-bit A-matrix 16x32 striping: lane = M%16; half0 lanes carry
  // K {0..7,16..23}, half1 lanes K {8..15,24..31} -> two 16B LDS reads.
  auto fragA = [&](int buf, int i) -> v16bf {
    const bf16_t* p = &sA[buf][wmB + i * 16 + lmod][0];
    v8bf lo = *(const v8bf*)(p + lhalf * 8);
    v8bf hi = *(const v8bf*)(p + 16 + lhalf * 8);
    return cat8(lo, hi);
  };
  // B-matrix 32x16: lane = N%16; half0 lanes K {0..7,16..23}, half1 lanes K {8..15,24..31}.
  auto fragB = [&](int buf, int j) -> v16bf {
    const bf16_t* p = &sB[buf][wnB + j * 16 + lmod][0];
    v8bf lo = *(const v8bf*)(p + lhalf * 8);
    v8bf hi = *(const v8bf*)(p + 16 + lhalf * 8);
    return cat8(lo, hi);
  };

  // prologue: stage tile 0
#if HAVE_ASYNC_LDS
  loadTileAsync(0, 0);
  waitAsync0();
#else
  {
    v8bf ga[4], gb[4];
    loadG(0, ga, gb);
    storeL(0, ga, gb);
  }
#endif
  __syncthreads();

  for (int kk = 0; kk < kEnd; kk += BK) {
    const int buf = (kk / BK) & 1;
    const bool pf = (kk + BK) < kEnd;

#if HAVE_ASYNC_LDS
    if (pf) loadTileAsync(kk + BK, buf ^ 1);   // overlaps with WMMAs below
#else
    v8bf ga[4], gb[4];
    if (pf) loadG(kk + BK, ga, gb);            // issue loads, consume later
#endif

    v16bf af[4];
    #pragma unroll
    for (int i = 0; i < 4; ++i) af[i] = fragA(buf, i);

    #pragma unroll
    for (int j = 0; j < 4; ++j) {
      v16bf bfg = fragB(buf, j);
      #pragma unroll
      for (int i = 0; i < 4; ++i)
        acc[i][j] = __builtin_amdgcn_wmma_f32_16x16x32_bf16(
            false, af[i], false, bfg, (short)0, acc[i][j], false, false);
    }

#if HAVE_ASYNC_LDS
    if (pf) waitAsync0();                      // LDS writes landed
#else
    if (pf) storeL(buf ^ 1, ga, gb);           // load-wait sits AFTER WMMAs
#endif
    __syncthreads();
  }

  // Epilogue. C/D layout: VGPR v -> row = v + 8*(lane/16), col = lane%16.
  const size_t cb = (size_t)bz * cBatch;
  #pragma unroll
  for (int i = 0; i < 4; ++i)
    #pragma unroll
    for (int j = 0; j < 4; ++j) {
      int row0 = m0 + wmB + i * 16 + lhalf * 8;
      int col  = n0 + wnB + j * 16 + lmod;
      #pragma unroll
      for (int v = 0; v < 8; ++v) {
        float val = acc[i][j][v];
        size_t idx = cb + (size_t)(row0 + v) * N + col;
        if (OUT_BF16) Cb[idx] = (bf16_t)val;
        else          Cf[idx] = val;
      }
    }
}

// bf16 32x32 LDS transpose (per-batch square matrices)
__global__ __launch_bounds__(256) void transpose_bf16_kernel(
    const bf16_t* __restrict__ in, bf16_t* __restrict__ out, int rows, int cols) {
  __shared__ bf16_t tile[32][34];
  const size_t bo = (size_t)blockIdx.z * rows * cols;
  in += bo; out += bo;
  int c0 = blockIdx.x * 32, r0 = blockIdx.y * 32;
  int tx = threadIdx.x & 31, ty = threadIdx.x >> 5;   // 32 x 8
  #pragma unroll
  for (int rr = ty; rr < 32; rr += 8)
    tile[rr][tx] = in[(size_t)(r0 + rr) * cols + c0 + tx];
  __syncthreads();
  #pragma unroll
  for (int rr = ty; rr < 32; rr += 8)
    out[(size_t)(c0 + rr) * rows + r0 + tx] = tile[tx][rr];
}

// causal softmax over one score row, staged once through LDS:
// P[q,j] = softmax(S[q,0..q]/scale), 0 past q
#define SEQ_MAX 2048
__global__ __launch_bounds__(256) void softmax_causal_kernel(
    const float* __restrict__ S, bf16_t* __restrict__ P, int seq, float invScale) {
  const int row = blockIdx.x;                 // b*seq + q
  const int q   = row % seq;
  const float* srow = S + (size_t)row * seq;
  bf16_t*      prow = P + (size_t)row * seq;
  const int L = q + 1;
  __shared__ float rowbuf[SEQ_MAX];
  __shared__ float red[256];
  const int t = threadIdx.x;

  float m = -INFINITY;
  for (int j = t; j < L; j += 256) {
    float l = srow[j] * invScale;             // single global read of the row
    rowbuf[j] = l;
    m = fmaxf(m, l);
  }
  red[t] = m; __syncthreads();
  for (int s = 128; s > 0; s >>= 1) {
    if (t < s) red[t] = fmaxf(red[t], red[t + s]);
    __syncthreads();
  }
  m = red[0]; __syncthreads();

  float sum = 0.f;
  for (int j = t; j < L; j += 256) sum += __expf(rowbuf[j] - m);
  red[t] = sum; __syncthreads();
  for (int s = 128; s > 0; s >>= 1) {
    if (t < s) red[t] += red[t + s];
    __syncthreads();
  }
  const float inv = 1.0f / red[0];

  for (int j = t; j < seq; j += 256) {
    float p = (j < L) ? __expf(rowbuf[j] - m) * inv : 0.0f;
    prow[j] = (bf16_t)p;
  }
}

extern "C" void kernel_launch(void* const* d_in, const int* in_sizes, int n_in,
                              void* d_out, int out_size, void* d_ws, size_t ws_size,
                              hipStream_t stream) {
  const float* x  = (const float*)d_in[0];
  const float* Wk = (const float*)d_in[1];
  const float* Wq = (const float*)d_in[2];
  const float* Wv = (const float*)d_in[3];
  float* out = (float*)d_out;

  const int Bb = 4, S = 2048, E = 2048, D = 2048;
  const size_t MS   = (size_t)Bb * S;            // 8192 flattened rows
  const size_t SD   = (size_t)S * D;             // per-batch QKV stride
  const size_t SS   = (size_t)S * S;             // per-batch score stride
  const size_t BF_X = MS * E * sizeof(bf16_t);   // 33.55 MB

  // Workspace layout (bytes):
  //  [0,       X)   Xbf   -> reused as Pbf after projections
  //  [X,     2X)    Qbf
  //  [2X,    3X)    Kbf
  //  [3X,    4X)    Vbf
  //  [4X,    5X)    Vt (V transposed per batch)
  //  [5X, 5X+67MB)  phase1: Wk/Wq/Wv bf16 (25MB); phase2+: scores f32 (67MB)
  const size_t need = 5 * BF_X + (size_t)Bb * SS * sizeof(float);
  if (ws_size < need) return;  // workspace too small; deterministic no-op

  char* ws = (char*)d_ws;
  bf16_t* Xbf  = (bf16_t*)(ws);
  bf16_t* Qbf  = (bf16_t*)(ws + 1 * BF_X);
  bf16_t* Kbf  = (bf16_t*)(ws + 2 * BF_X);
  bf16_t* Vbf  = (bf16_t*)(ws + 3 * BF_X);
  bf16_t* Vt   = (bf16_t*)(ws + 4 * BF_X);
  char*   regB = ws + 5 * BF_X;
  bf16_t* Wkbf = (bf16_t*)(regB);
  bf16_t* Wqbf = (bf16_t*)(regB + (size_t)D * E * sizeof(bf16_t));
  bf16_t* Wvbf = (bf16_t*)(regB + 2 * (size_t)D * E * sizeof(bf16_t));
  float*  Sf   = (float*)(regB);               // overlays W region (phase 2+)
  bf16_t* Pbf  = Xbf;                          // reuse X region for probs

  // 1) fp32 -> bf16 conversions
  {
    int n8 = (int)(MS * E / 8);
    cvt_f32_bf16_kernel<<<(n8 + 255) / 256, 256, 0, stream>>>(x, Xbf, n8);
    n8 = (int)((size_t)D * E / 8);
    cvt_f32_bf16_kernel<<<(n8 + 255) / 256, 256, 0, stream>>>(Wk, Wkbf, n8);
    cvt_f32_bf16_kernel<<<(n8 + 255) / 256, 256, 0, stream>>>(Wq, Wqbf, n8);
    cvt_f32_bf16_kernel<<<(n8 + 255) / 256, 256, 0, stream>>>(Wv, Wvbf, n8);
  }

  // 2) projections: [8192,2048] = Xbf[8192,2048] @ W[2048,2048]^T  (bf16 out)
  {
    dim3 grid(D / BN, (unsigned)(MS / BM), 1);
    gemm_nt_kernel<1, 0><<<grid, 128, 0, stream>>>(
        Xbf, Wqbf, nullptr, Qbf, (int)MS, D, E, 0, 0, 0);
    gemm_nt_kernel<1, 0><<<grid, 128, 0, stream>>>(
        Xbf, Wkbf, nullptr, Kbf, (int)MS, D, E, 0, 0, 0);
    gemm_nt_kernel<1, 0><<<grid, 128, 0, stream>>>(
        Xbf, Wvbf, nullptr, Vbf, (int)MS, D, E, 0, 0, 0);
  }

  // 3) transpose V per batch: Vt[b][d][s] = V[b][s][d]
  transpose_bf16_kernel<<<dim3(D / 32, S / 32, Bb), 256, 0, stream>>>(Vbf, Vt, S, D);

  // 4) scores: Sf[b] = Q[b] @ K[b]^T  (f32 out, skip fully-masked blocks)
  gemm_nt_kernel<0, 1><<<dim3(S / BN, S / BM, Bb), 128, 0, stream>>>(
      Qbf, Kbf, Sf, nullptr, S, S, D, SD, SD, SS);

  // 5) causal softmax (scale = 1/sqrt(D)) -> bf16 probabilities
  softmax_causal_kernel<<<(unsigned)(Bb * S), 256, 0, stream>>>(
      Sf, Pbf, S, 1.0f / sqrtf((float)D));

  // 6) output: out[b] = P[b] @ V[b]  via Vt (NT form), K capped at diagonal
  gemm_nt_kernel<0, 2><<<dim3(D / BN, S / BM, Bb), 128, 0, stream>>>(
      Pbf, Vt, out, nullptr, S, D, S, SS, SD, SD);
}
